// MyAttention_31207232372885
// MI455X (gfx1250) — matrix operand order repaired and checked by reading, not compiled, not run on hardware
//
#include <hip/hip_runtime.h>
#include <hip/hip_bf16.h>
#include <cstdint>

typedef _Float16 v16h __attribute__((ext_vector_type(16)));
typedef _Float16 v8h  __attribute__((ext_vector_type(8)));
typedef float    v8f  __attribute__((ext_vector_type(8)));
typedef unsigned int u32x4 __attribute__((ext_vector_type(4)));
typedef int          i32x4 __attribute__((ext_vector_type(4)));
typedef int          i32x8 __attribute__((ext_vector_type(8)));

#define B_   2
#define S_   2048
#define H_   2048
#define NH_  16
#define KVH_ 4
#define HD_  128
#define BS_  (B_ * S_)

__device__ __forceinline__ v16h make_v16(v8h a, v8h b) {
    v16h r;
#pragma unroll
    for (int i = 0; i < 8; ++i) { r[i] = a[i]; r[i + 8] = b[i]; }
    return r;
}

__device__ __forceinline__ v8f zero_v8f() {
    v8f z = {0.f, 0.f, 0.f, 0.f, 0.f, 0.f, 0.f, 0.f};
    return z;
}

// ---- Tensor Data Mover: 2D tile (elements are 2 bytes). Issued once per wave.
// D# layout per CDNA5 ISA.
__device__ __forceinline__ void tdm_load_2d(unsigned lds_off, const void* gaddr,
                                            unsigned tensor_d0, unsigned tensor_d1,
                                            unsigned tile_d0, unsigned tile_d1,
                                            unsigned stride0_elems) {
    unsigned long long ga = (unsigned long long)(uintptr_t)gaddr;
    u32x4 g0;
    g0[0] = 1u;                                            // count=1 (valid user D#)
    g0[1] = lds_off;                                       // lds_addr (bytes)
    g0[2] = (unsigned)(ga & 0xFFFFFFFFu);                  // global_addr[31:0]
    g0[3] = (unsigned)((ga >> 32) & 0x01FFFFFFu) | (2u << 30);  // addr[56:32] | type=2
    i32x8 g1;
    g1[0] = (int)(1u << 16);                               // data_size=1 -> 2 bytes
    g1[1] = (int)((tensor_d0 & 0xFFFFu) << 16);            // tensor_dim0[15:0] @ bits63:48
    g1[2] = (int)(((tensor_d0 >> 16) & 0xFFFFu) | ((tensor_d1 & 0xFFFFu) << 16));
    g1[3] = (int)(((tensor_d1 >> 16) & 0xFFFFu) | ((tile_d0 & 0xFFFFu) << 16));
    g1[4] = (int)(tile_d1 & 0xFFFFu);                      // tile_dim1; tile_dim2=0
    g1[5] = (int)stride0_elems;                            // tensor_dim0_stride[31:0]
    g1[6] = 0;
    g1[7] = 0;
    i32x4 z4 = {0, 0, 0, 0};
#if __clang_major__ >= 23
    i32x8 z8 = {0, 0, 0, 0, 0, 0, 0, 0};
    __builtin_amdgcn_tensor_load_to_lds(g0, g1, z4, z4, z8, 0);
#else
    __builtin_amdgcn_tensor_load_to_lds(g0, g1, z4, z4, 0);
#endif
}

// ---------------- elementwise conversion kernels ----------------

__global__ void cast_f32_f16_kernel(const float* __restrict__ src,
                                    _Float16* __restrict__ dst, int n) {
    int i = blockIdx.x * blockDim.x + threadIdx.x;
    if (i < n) dst[i] = (_Float16)src[i];
}

// dst[n*K + k] = src[k*N + n]  (weight K x N -> transposed N x K, f16)
__global__ void transpose_cast_kernel(const float* __restrict__ src,
                                      _Float16* __restrict__ dst, int K, int N) {
    int i = blockIdx.x * blockDim.x + threadIdx.x;
    if (i >= K * N) return;
    int n = i / K;
    int k = i - n * K;
    dst[i] = (_Float16)src[k * N + n];
}

// RoPE + cast. src/dst layout [B*S, nheads, 128]
__global__ void rope_cast_kernel(const float* __restrict__ src,
                                 _Float16* __restrict__ dst,
                                 int nheads, int total) {
    int i = blockIdx.x * blockDim.x + threadIdx.x;
    if (i >= total) return;
    int d   = i & (HD_ - 1);
    int row = i / (nheads * HD_);      // b*S + s
    int pos = row & (S_ - 1);          // s
    int fi  = d & 63;
    float inv = __powf(10000.0f, -(float)fi * (1.0f / 64.0f));
    float ang = (float)pos * inv;
    float sn, cs;
    __sincosf(ang, &sn, &cs);
    float x = src[i];
    float rot = (d < 64) ? -src[i + 64] : src[i - 64];
    dst[i] = (_Float16)(x * cs + rot * sn);
}

// v [B*S, KVH, HD] f32 -> vt [B, KVH, HD, S] f16
__global__ void v_transpose_cast_kernel(const float* __restrict__ src,
                                        _Float16* __restrict__ dst, int total) {
    int i = blockIdx.x * blockDim.x + threadIdx.x;
    if (i >= total) return;
    int s   = i & (S_ - 1);
    int d   = (i >> 11) & (HD_ - 1);
    int kvh = (i >> 18) & (KVH_ - 1);
    int b   = i >> 20;
    dst[i] = (_Float16)src[(((size_t)(b * S_ + s) * KVH_) + kvh) * HD_ + d];
}

// ---------------- WMMA GEMM: C(MxN,f32) = X(MxK,f16) @ WT(NxK,f16)^T ----------------
// block = 128 threads (4 waves); wave w -> 32x64 output strip; block tile 128x64.
// B fragments reused across two A fragments: 8 WMMAs per 6 fragment loads / k-step.

__global__ void gemm_wmma_kernel(const _Float16* __restrict__ X,
                                 const _Float16* __restrict__ WT,
                                 float* __restrict__ C,
                                 int M, int N, int K) {
    int lane = threadIdx.x & 31;
    int wid  = threadIdx.x >> 5;
    int lr   = lane & 15;
    int hi   = lane >> 4;
    int m0   = blockIdx.y * 128 + wid * 32;
    int n0   = blockIdx.x * 64;

    v8f acc[2][4];
#pragma unroll
    for (int i = 0; i < 2; ++i)
#pragma unroll
        for (int t = 0; t < 4; ++t) acc[i][t] = zero_v8f();

    const _Float16* xrow0 = X + (size_t)(m0 + lr) * K;
    const _Float16* xrow1 = X + (size_t)(m0 + 16 + lr) * K;
    for (int k0 = 0; k0 < K; k0 += 32) {
        __builtin_prefetch(xrow0 + k0 + 1024, 0, 1);
        v16h a0 = make_v16(*(const v8h*)(xrow0 + k0 + hi * 16),
                           *(const v8h*)(xrow0 + k0 + hi * 16 + 8));
        v16h a1 = make_v16(*(const v8h*)(xrow1 + k0 + hi * 16),
                           *(const v8h*)(xrow1 + k0 + hi * 16 + 8));
#pragma unroll
        for (int t = 0; t < 4; ++t) {
            const _Float16* wrow = WT + (size_t)(n0 + t * 16 + lr) * K + k0 + hi * 16;
            v16h b = *(const v16h*)wrow;
            acc[0][t] = __builtin_amdgcn_wmma_f32_16x16x32_f16(
                false, a0, false, b, (short)0, acc[0][t], false, false);
            acc[1][t] = __builtin_amdgcn_wmma_f32_16x16x32_f16(
                false, a1, false, b, (short)0, acc[1][t], false, false);
        }
    }
#pragma unroll
    for (int i = 0; i < 2; ++i)
#pragma unroll
        for (int t = 0; t < 4; ++t)
#pragma unroll
            for (int r = 0; r < 8; ++r)
                C[(size_t)(m0 + i * 16 + r + hi * 8) * N + n0 + t * 16 + lr] = acc[i][t][r];
}

// ---------------- flash attention ----------------
// grid: B * 4(kv head groups) * 128(q tiles); block: 128 threads = 4 waves,
// wave w handles head h = hg*4 + w. All waves share qt -> identical trip counts,
// and share the same KV head -> K/V^T tiles staged once per block via TDM.

__global__ void flash_attn_kernel(const _Float16* __restrict__ qh,  // [B,S,NH,HD]
                                  const _Float16* __restrict__ kh,  // [B,S,KVH,HD]
                                  const _Float16* __restrict__ vt,  // [B,KVH,HD,S]
                                  _Float16* __restrict__ ctx) {     // [B,S,NH,HD]
    const float scale = 0.08838834764831845f;  // 1/sqrt(128)
    int lane = threadIdx.x & 31;
    int wid  = threadIdx.x >> 5;
    int lr   = lane & 15;
    int hi   = lane >> 4;

    int idx = blockIdx.x;
    int qt  = idx & 127;
    int hg  = (idx >> 7) & 3;
    int b   = idx >> 9;
    int h   = hg * 4 + wid;
    int kvh = hg;

    __shared__ __align__(16) _Float16 Kt[32 * 128];   // [kv 32][d 128]
    __shared__ __align__(16) _Float16 Vt[128 * 32];   // [d 128][kv 32]
    __shared__ __align__(16) _Float16 P[4][16 * 32];
    _Float16* Pw = P[wid];
    unsigned Kt_off = (unsigned)(uintptr_t)Kt;        // LDS byte offset (addr[31:0])
    unsigned Vt_off = (unsigned)(uintptr_t)Vt;

    // Q fragments: 4 chunks of K=32 along HD
    v16h qf[4];
    const _Float16* qbase = qh + ((size_t)(b * S_ + qt * 16 + lr) * NH_ + h) * HD_;
#pragma unroll
    for (int c = 0; c < 4; ++c) {
        qf[c] = make_v16(*(const v8h*)(qbase + c * 32 + hi * 16),
                         *(const v8h*)(qbase + c * 32 + hi * 16 + 8));
    }

    v8f acc[8];
#pragma unroll
    for (int t = 0; t < 8; ++t) acc[t] = zero_v8f();
    float mrow[8], lrow[8];
#pragma unroll
    for (int r = 0; r < 8; ++r) { mrow[r] = -1e30f; lrow[r] = 0.f; }

    const _Float16* kbb = kh + ((size_t)b * S_ * KVH_ + kvh) * HD_;
    const _Float16* vbb = vt + (size_t)(b * KVH_ + kvh) * HD_ * S_;

    int kv_end = qt * 16 + 16;
    for (int kv0 = 0; kv0 < kv_end; kv0 += 32) {
        // ---- stage K (32x128) and V^T (128x32) tiles into LDS via TDM ----
        if (wid == 0) {
            tdm_load_2d(Kt_off, kbb + (size_t)kv0 * (KVH_ * HD_),
                        /*tensor_d0=*/HD_, /*tensor_d1=*/32,
                        /*tile_d0=*/HD_, /*tile_d1=*/32, /*stride0=*/KVH_ * HD_);
            tdm_load_2d(Vt_off, vbb + kv0,
                        /*tensor_d0=*/32, /*tensor_d1=*/HD_,
                        /*tile_d0=*/32, /*tile_d1=*/HD_, /*stride0=*/S_);
            __builtin_amdgcn_s_wait_tensorcnt((short)0);
        }
        __syncthreads();

        // ---- scores: 16 q rows x 32 kv cols (fragments from LDS) ----
        v8f s0 = zero_v8f(), s1 = zero_v8f();
#pragma unroll
        for (int c = 0; c < 4; ++c) {
            v16h b0 = *(const v16h*)(&Kt[lr * HD_ + c * 32 + hi * 16]);
            v16h b1 = *(const v16h*)(&Kt[(16 + lr) * HD_ + c * 32 + hi * 16]);
            s0 = __builtin_amdgcn_wmma_f32_16x16x32_f16(false, qf[c], false, b0,
                                                        (short)0, s0, false, false);
            s1 = __builtin_amdgcn_wmma_f32_16x16x32_f16(false, qf[c], false, b1,
                                                        (short)0, s1, false, false);
        }
        // ---- online softmax update ----
#pragma unroll
        for (int r = 0; r < 8; ++r) {
            int row = qt * 16 + r + hi * 8;
            float v0 = s0[r] * scale;
            float v1 = s1[r] * scale;
            if (kv0 + lr > row)      v0 = -1e30f;
            if (kv0 + 16 + lr > row) v1 = -1e30f;
            float mx = fmaxf(v0, v1);
            mx = fmaxf(mx, __shfl_xor(mx, 1, 32));
            mx = fmaxf(mx, __shfl_xor(mx, 2, 32));
            mx = fmaxf(mx, __shfl_xor(mx, 4, 32));
            mx = fmaxf(mx, __shfl_xor(mx, 8, 32));
            float mnew  = fmaxf(mrow[r], mx);
            float alpha = __expf(mrow[r] - mnew);
            float p0 = __expf(v0 - mnew);
            float p1 = __expf(v1 - mnew);
            float sum = p0 + p1;
            sum += __shfl_xor(sum, 1, 32);
            sum += __shfl_xor(sum, 2, 32);
            sum += __shfl_xor(sum, 4, 32);
            sum += __shfl_xor(sum, 8, 32);
            lrow[r] = lrow[r] * alpha + sum;
            mrow[r] = mnew;
#pragma unroll
            for (int dt = 0; dt < 8; ++dt) acc[dt][r] *= alpha;
            Pw[(r + hi * 8) * 32 + lr]      = (_Float16)p0;
            Pw[(r + hi * 8) * 32 + 16 + lr] = (_Float16)p1;
        }
        __syncthreads();
        // C-layout -> A-layout through LDS
        v16h pf = make_v16(*(const v8h*)(&Pw[lr * 32 + hi * 16]),
                           *(const v8h*)(&Pw[lr * 32 + hi * 16 + 8]));
        // ---- ctx += P @ V (V^T fragments from LDS) ----
#pragma unroll
        for (int dt = 0; dt < 8; ++dt) {
            v16h bf = *(const v16h*)(&Vt[(dt * 16 + lr) * 32 + hi * 16]);
            acc[dt] = __builtin_amdgcn_wmma_f32_16x16x32_f16(false, pf, false, bf,
                                                             (short)0, acc[dt], false, false);
        }
        __syncthreads();  // all waves done with Kt/Vt/P before next stage
    }
    // ---- epilogue: normalize & store f16 ctx ----
#pragma unroll
    for (int r = 0; r < 8; ++r) {
        float invl = 1.0f / lrow[r];
        size_t obase = ((size_t)(b * S_ + qt * 16 + r + hi * 8) * NH_ + h) * HD_;
#pragma unroll
        for (int dt = 0; dt < 8; ++dt) {
            ctx[obase + dt * 16 + lr] = (_Float16)(acc[dt][r] * invl);
        }
    }
}

// ---------------- launch ----------------

extern "C" void kernel_launch(void* const* d_in, const int* in_sizes, int n_in,
                              void* d_out, int out_size, void* d_ws, size_t ws_size,
                              hipStream_t stream) {
    const float* hidden = (const float*)d_in[0];  // [B,S,H]
    const float* Wq = (const float*)d_in[1];      // [H, NH*HD]
    const float* Wk = (const float*)d_in[2];      // [H, KVH*HD]
    const float* Wv = (const float*)d_in[3];      // [H, KVH*HD]
    const float* Wo = (const float*)d_in[4];      // [NH*HD, H]
    float* out = (float*)d_out;                   // [B,S,H]

    const int QN = NH_ * HD_;   // 2048
    const int KN = KVH_ * HD_;  // 512

    size_t off = 0;
    auto alloc = [&](size_t bytes) {
        void* p = (char*)d_ws + off;
        off += (bytes + 255) & ~(size_t)255;
        return p;
    };
    _Float16* hid_h = (_Float16*)alloc((size_t)BS_ * H_ * 2);
    _Float16* WqT   = (_Float16*)alloc((size_t)QN * H_ * 2);
    _Float16* WkT   = (_Float16*)alloc((size_t)KN * H_ * 2);
    _Float16* WvT   = (_Float16*)alloc((size_t)KN * H_ * 2);
    _Float16* WoT   = (_Float16*)alloc((size_t)H_ * QN * 2);
    float* q32      = (float*)alloc((size_t)BS_ * QN * 4);
    float* k32      = (float*)alloc((size_t)BS_ * KN * 4);
    float* v32      = (float*)alloc((size_t)BS_ * KN * 4);
    _Float16* q_h   = (_Float16*)alloc((size_t)BS_ * QN * 2);
    _Float16* k_h   = (_Float16*)alloc((size_t)BS_ * KN * 2);
    _Float16* vt_h  = (_Float16*)alloc((size_t)BS_ * KN * 2);
    _Float16* ctx_h = (_Float16*)(void*)q32;  // reuse: q32 dead after rope_cast

    // 1. casts / transposes
    {
        int n = BS_ * H_;
        cast_f32_f16_kernel<<<(n + 255) / 256, 256, 0, stream>>>(hidden, hid_h, n);
    }
    {
        int n = H_ * QN;
        transpose_cast_kernel<<<(n + 255) / 256, 256, 0, stream>>>(Wq, WqT, H_, QN);
        transpose_cast_kernel<<<(n + 255) / 256, 256, 0, stream>>>(Wo, WoT, QN, H_);
    }
    {
        int n = H_ * KN;
        transpose_cast_kernel<<<(n + 255) / 256, 256, 0, stream>>>(Wk, WkT, H_, KN);
        transpose_cast_kernel<<<(n + 255) / 256, 256, 0, stream>>>(Wv, WvT, H_, KN);
    }
    // 2. QKV projections (WMMA)
    gemm_wmma_kernel<<<dim3(QN / 64, BS_ / 128), 128, 0, stream>>>(hid_h, WqT, q32, BS_, QN, H_);
    gemm_wmma_kernel<<<dim3(KN / 64, BS_ / 128), 128, 0, stream>>>(hid_h, WkT, k32, BS_, KN, H_);
    gemm_wmma_kernel<<<dim3(KN / 64, BS_ / 128), 128, 0, stream>>>(hid_h, WvT, v32, BS_, KN, H_);
    // 3. RoPE + cast, V transpose-cast
    {
        int n = BS_ * QN;
        rope_cast_kernel<<<(n + 255) / 256, 256, 0, stream>>>(q32, q_h, NH_, n);
    }
    {
        int n = BS_ * KN;
        rope_cast_kernel<<<(n + 255) / 256, 256, 0, stream>>>(k32, k_h, KVH_, n);
        v_transpose_cast_kernel<<<(n + 255) / 256, 256, 0, stream>>>(v32, vt_h, n);
    }
    // 4. causal flash attention (WMMA + TDM staging)
    flash_attn_kernel<<<B_ * 4 * (S_ / 16), 128, 0, stream>>>(q_h, k_h, vt_h, ctx_h);
    // 5. output projection (WMMA) -> fp32 out
    gemm_wmma_kernel<<<dim3(H_ / 64, BS_ / 128), 128, 0, stream>>>(ctx_h, WoT, out, BS_, H_, QN);
}
